// E8SparseMoETriality_67370857005587
// MI455X (gfx1250) — matrix operand order repaired and checked by reading, not compiled, not run on hardware
//
#include <hip/hip_runtime.h>

// ---------------- types for CDNA5 WMMA ----------------
typedef __bf16 v16bf __attribute__((ext_vector_type(16)));
typedef float  v8f   __attribute__((ext_vector_type(8)));

#define DIM 240
#define NEXP 8
#define DEPTH 8
#define BB 8
#define SS 2048
#define NTOK (BB * SS)          // 16384
#define NTTILE (NTOK / 16)      // 1024 token tiles of 16
#define NNTILE (DIM / 16)       // 15
#define NKTILE 8                // K padded 240 -> 256, tiles of 32
#define NSUPER 64               // token supergroups of 256 tokens (8 waves x 32)

// ---- workspace layout (bytes) ----
static constexpr size_t WS_COMBINE = 0;                               // 16384*8*4 = 512KB
static constexpr size_t WS_HA      = 524288;
static constexpr size_t WS_HB      = WS_HA + (size_t)NTOK * DIM * 4;
static constexpr size_t WS_AFRAG   = WS_HB + (size_t)NTOK * DIM * 4;
static constexpr size_t WS_BFRAG   = WS_AFRAG + (size_t)NTTILE * NKTILE * 512 * 2;
// bfrag: [layer][ntile][e][ktile][lane(32)][elem(16)] bf16; layer stride (ushorts):
static constexpr size_t BFRAG_LAYER_STRIDE = (size_t)NNTILE * NEXP * NKTILE * 512; // 491520

__device__ __forceinline__ unsigned short f2bf(float f) {
    unsigned u = __builtin_bit_cast(unsigned, f);
    unsigned r = u + 0x7FFFu + ((u >> 16) & 1u);   // round-to-nearest-even
    return (unsigned short)(r >> 16);
}

// ---------------- 0. pre-swizzle expert weights fp32 -> bf16 B-fragments ----------------
// B 32x16 bf16 fragment layout (ISA 7.12.2): lane -> N = lane%16,
// element i -> K = (i<8 ? i : 16+(i-8)) + 8*(lane>=16), within each 32-K tile.
__global__ __launch_bounds__(256) void packW_kernel(const float* __restrict__ eW,
                                                    unsigned short* __restrict__ bfrag) {
    int blk   = blockIdx.x;            // (l*8+e)*15 + ntile, 0..959
    int ntile = blk % NNTILE;
    int le    = blk / NNTILE;          // l*8 + e
    int l     = le >> 3;
    int e     = le & 7;
    int kt    = threadIdx.x >> 5;
    int lane  = threadIdx.x & 31;
    int lh    = lane >> 4;
    int n     = ntile * 16 + (lane & 15);
    const float* W = eW + (size_t)le * DIM * DIM;               // [K=240][N=240]
    unsigned short* dst = bfrag +
        (((((size_t)l * NNTILE + ntile) * NEXP + e) * NKTILE + kt) * 32 + lane) * 16;
#pragma unroll
    for (int i = 0; i < 16; ++i) {
        int K = kt * 32 + ((i < 8) ? i : (16 + (i - 8))) + 8 * lh;
        float v = (K < DIM) ? W[(size_t)K * DIM + n] : 0.0f;
        dst[i] = f2bf(v);
    }
}

// ---------------- 1. cycle block ----------------
__global__ __launch_bounds__(256) void cycle_kernel(const float* __restrict__ x,
                                                    const int* __restrict__ stepp,
                                                    const float* __restrict__ roots,
                                                    const float* __restrict__ projW,
                                                    float* __restrict__ h) {
    __shared__ float sh[DIM];
    int d = threadIdx.x;               // 240 threads
    int t = blockIdx.x;                // token
    int s = t % SS;
    int pos = s % 240;
    int c = d % 80;
    float emb = 0.f;
#pragma unroll
    for (int j = 0; j < 8; ++j) emb += roots[pos * 8 + j] * projW[j * 80 + c];
    float pump = 0.8f * sinf((float)(*stepp) * 0.006f * 2.0f * 3.14159265358979f);
    float xv = x[(size_t)t * DIM + d];
    float ce = cosf(emb), se = sinf(emb);
    float x1 = xv * (ce + pump);
    sh[d] = x1;
    __syncthreads();
    float x2 = sh[(d + DIM - 1) % DIM] * se;
    __syncthreads();
    sh[d] = x2;
    __syncthreads();
    float x3 = sh[(d + DIM - 1) % DIM] * ce;
    h[(size_t)t * DIM + d] = (x1 + x2 + x3) * (1.0f / 3.0f);
}

// ---------------- 2. gating (fp32, one thread per token) ----------------
__global__ __launch_bounds__(64) void gate_kernel(const float* __restrict__ h,
                                                  const float* __restrict__ gW,   // [240][8]
                                                  const float* __restrict__ gb,   // [8]
                                                  float* __restrict__ combine) {  // [NTOK][8]
    __shared__ float wsh[DIM * NEXP];
    __shared__ float bsh[NEXP];
    int tid = threadIdx.x;
    for (int i = tid; i < DIM * NEXP; i += 64) wsh[i] = gW[i];
    if (tid < NEXP) bsh[tid] = gb[tid];
    __syncthreads();
    int t = blockIdx.x * 64 + tid;
    const float* hp = h + (size_t)t * DIM;
    float lg[NEXP];
#pragma unroll
    for (int e = 0; e < NEXP; ++e) lg[e] = bsh[e];
    for (int d = 0; d < DIM; ++d) {
        float hv = hp[d];
#pragma unroll
        for (int e = 0; e < NEXP; ++e) lg[e] += hv * wsh[d * NEXP + e];
    }
    float mx = lg[0];
#pragma unroll
    for (int e = 1; e < NEXP; ++e) mx = fmaxf(mx, lg[e]);
    float p[NEXP], sum = 0.f;
#pragma unroll
    for (int e = 0; e < NEXP; ++e) { p[e] = expf(lg[e] - mx); sum += p[e]; }
    float inv = 1.0f / sum;
#pragma unroll
    for (int e = 0; e < NEXP; ++e) p[e] *= inv;
    // top-2, jax tie semantics: earliest index wins on ties
    int e1 = 0;
#pragma unroll
    for (int e = 1; e < NEXP; ++e) if (p[e] > p[e1]) e1 = e;
    int e2 = (e1 == 0) ? 1 : 0;
#pragma unroll
    for (int e = 0; e < NEXP; ++e) if (e != e1 && p[e] > p[e2]) e2 = e;
    float rs = 1.0f / (p[e1] + p[e2]);
    float w1 = p[e1] * rs, w2 = p[e2] * rs;
    float* cp = combine + (size_t)t * NEXP;
#pragma unroll
    for (int e = 0; e < NEXP; ++e) cp[e] = (e == e1) ? w1 : ((e == e2) ? w2 : 0.0f);
}

// ---------------- 3. pack activations fp32 -> bf16 A-fragments ----------------
// A 16x32 bf16 layout (ISA 7.12.2): lane -> M = lane%16; element i (vgpr v=i/2, half=i%2):
//   v<4 : K = 2v+half + 8*(lane>=16);  v>=4 : K = 16 + 2(v-4)+half + 8*(lane>=16)
__global__ __launch_bounds__(256) void packA_kernel(const float* __restrict__ h,
                                                    unsigned short* __restrict__ afrag) {
    int ttile = blockIdx.x;            // 0..1023
    int kt    = threadIdx.x >> 5;
    int lane  = threadIdx.x & 31;
    int m     = lane & 15;
    int lh    = lane >> 4;
    unsigned short* dst = afrag + (((size_t)ttile * NKTILE + kt) * 32 + lane) * 16;
    const float* hp = h + (size_t)(ttile * 16 + m) * DIM;
#pragma unroll
    for (int i = 0; i < 16; ++i) {
        int v = i >> 1, hf = i & 1;
        int k = ((v < 4) ? (2 * v + hf) : (16 + 2 * (v - 4) + hf)) + 8 * lh + kt * 32;
        dst[i] = f2bf((k < DIM) ? hp[k] : 0.0f);
    }
}

// ---------------- 4. dense MoE GEMM via v_wmma_f32_16x16x32_bf16 ----------------
// 8 waves / 256 tokens per workgroup share one ntile's B fragments through LDS
// (staged in two 32KB halves of 4 experts each) -> 8x less L2 weight traffic.
// All 4 B fragments of a K-step are loaded into distinct registers before the
// WMMAs so the ds_load latency pipelines instead of serializing.
__global__ __launch_bounds__(256) void moe_kernel(const unsigned short* __restrict__ afrag,
                                                  const unsigned short* __restrict__ bfragL,
                                                  const float* __restrict__ combine,
                                                  const float* __restrict__ ebL,   // [8][240]
                                                  float* __restrict__ hout) {
    __shared__ __align__(16) unsigned short ldsB[4 * NKTILE * 512]; // 4 experts = 32KB
    __shared__ float csh[256][NEXP];                                // 8KB

    int blk    = blockIdx.x;           // 0..(64*15-1)
    int ntile  = blk % NNTILE;
    int tsuper = blk / NNTILE;         // 0..63 : tokens tsuper*256 .. +255
    int tid    = threadIdx.x;
    int w      = tid >> 5;             // wave 0..7
    int lane   = tid & 31;
    int laneN  = lane & 15;
    int lh     = lane >> 4;

    // stage combine weights for this workgroup's 256 tokens
    {
        const float* cp = combine + (size_t)(tsuper * 256 + w * 32 + lane) * NEXP;
#pragma unroll
        for (int e = 0; e < NEXP; ++e) csh[w * 32 + lane][e] = cp[e];
    }

    int tgrp = tsuper * 8 + w;         // group of 2 token tiles owned by this wave
    const unsigned short* a0 = afrag + (((size_t)(tgrp * 2 + 0) * NKTILE) * 32 + lane) * 16;
    const unsigned short* a1 = afrag + (((size_t)(tgrp * 2 + 1) * NKTILE) * 32 + lane) * 16;
    // contiguous 64KB B region for this ntile: [e(8)][kt(8)][lane(32)][16]
    const unsigned short* bsrc = bfragL + (size_t)ntile * NEXP * NKTILE * 512;

    v8f acc[2][NEXP];
#pragma unroll
    for (int tt = 0; tt < 2; ++tt)
#pragma unroll
        for (int e = 0; e < NEXP; ++e) acc[tt][e] = (v8f){};

#pragma unroll
    for (int es = 0; es < 2; ++es) {
        __syncthreads();
        // cooperative copy of 4 experts (32KB = 2048 x uint4) into LDS
        {
            const uint4* src = (const uint4*)(bsrc + (size_t)es * 4 * NKTILE * 512);
            uint4* dst = (uint4*)ldsB;
            for (int i = tid; i < 2048; i += 256) dst[i] = src[i];
        }
        __syncthreads();
        for (int kt = 0; kt < NKTILE; ++kt) {
            // issue all loads for this K-step first, then the 8 WMMAs
            v16bf fa0 = *(const v16bf*)(a0 + (size_t)kt * 512);
            v16bf fa1 = *(const v16bf*)(a1 + (size_t)kt * 512);
            v16bf fb[4];
#pragma unroll
            for (int el = 0; el < 4; ++el)
                fb[el] = *(const v16bf*)(ldsB + (((size_t)el * NKTILE + kt) * 32 + lane) * 16);
#pragma unroll
            for (int el = 0; el < 4; ++el) {
                acc[0][es * 4 + el] = __builtin_amdgcn_wmma_f32_16x16x32_bf16(
                    false, fa0, false, fb[el], (short)0, acc[0][es * 4 + el], false, false);
                acc[1][es * 4 + el] = __builtin_amdgcn_wmma_f32_16x16x32_bf16(
                    false, fa1, false, fb[el], (short)0, acc[1][es * 4 + el], false, false);
            }
        }
    }

    // epilogue: out[tok][n] = sum_e c[tok][e] * (gemm_e + b_e[n])
    int n = ntile * 16 + laneN;
    float ebn[NEXP];
#pragma unroll
    for (int e = 0; e < NEXP; ++e) ebn[e] = ebL[e * DIM + n];
#pragma unroll
    for (int tt = 0; tt < 2; ++tt) {
#pragma unroll
        for (int r = 0; r < 8; ++r) {
            int m  = r + 8 * lh;               // C/D layout: vgpr r, lane-hi -> M=r+8
            int tl = tt * 16 + m;
            float v = 0.f;
#pragma unroll
            for (int e = 0; e < NEXP; ++e) v += csh[w * 32 + tl][e] * (acc[tt][e][r] + ebn[e]);
            hout[(size_t)(tgrp * 32 + tl) * DIM + n] = v;
        }
    }
}

// ---------------- 5. h += LayerNorm(h), in place ----------------
__global__ __launch_bounds__(256) void ln_kernel(float* __restrict__ h,
                                                 const float* __restrict__ gamma,
                                                 const float* __restrict__ beta) {
    __shared__ float red[256];
    int d = threadIdx.x;
    int t = blockIdx.x;
    float v = (d < DIM) ? h[(size_t)t * DIM + d] : 0.f;
    red[d] = v;
    __syncthreads();
    for (int o = 128; o > 0; o >>= 1) { if (d < o) red[d] += red[d + o]; __syncthreads(); }
    float mean = red[0] * (1.0f / DIM);
    __syncthreads();
    float dv = (d < DIM) ? (v - mean) : 0.f;
    red[d] = dv * dv;
    __syncthreads();
    for (int o = 128; o > 0; o >>= 1) { if (d < o) red[d] += red[d + o]; __syncthreads(); }
    float rstd = rsqrtf(red[0] * (1.0f / DIM) + 1e-5f);
    if (d < DIM) h[(size_t)t * DIM + d] = v + dv * rstd * gamma[d] + beta[d];
}

// ---------------- 6. mean-pool + head + sigmoid ----------------
__global__ __launch_bounds__(256) void head_kernel(const float* __restrict__ h,
                                                   const float* __restrict__ hW,
                                                   const float* __restrict__ hb,
                                                   float* __restrict__ out) {
    __shared__ float red[256];
    int b = blockIdx.x, tid = threadIdx.x;
    const float* hp = h + (size_t)b * SS * DIM;
    float acc = 0.f;
    for (int i = tid; i < SS * DIM; i += 256) acc += hp[i] * hW[i % DIM];
    red[tid] = acc;
    __syncthreads();
    for (int o = 128; o > 0; o >>= 1) { if (tid < o) red[tid] += red[tid + o]; __syncthreads(); }
    if (tid == 0) {
        float z = red[0] * (1.0f / SS) + hb[0];
        out[b] = 1.0f / (1.0f + expf(-z));
    }
}

extern "C" void kernel_launch(void* const* d_in, const int* in_sizes, int n_in,
                              void* d_out, int out_size, void* d_ws, size_t ws_size,
                              hipStream_t stream) {
    (void)in_sizes; (void)n_in; (void)out_size; (void)ws_size;
    const float* x     = (const float*)d_in[0];
    const int*   step  = (const int*)d_in[1];
    const float* roots = (const float*)d_in[2];
    const float* projW = (const float*)d_in[3];
    const float* gateW = (const float*)d_in[4];
    const float* gateB = (const float*)d_in[5];
    const float* eW    = (const float*)d_in[6];
    const float* eb    = (const float*)d_in[7];
    const float* gamma = (const float*)d_in[8];
    const float* beta  = (const float*)d_in[9];
    const float* headW = (const float*)d_in[10];
    const float* headB = (const float*)d_in[11];
    float* out = (float*)d_out;
    char*  ws  = (char*)d_ws;

    float* combine        = (float*)(ws + WS_COMBINE);
    float* hA             = (float*)(ws + WS_HA);
    float* hB             = (float*)(ws + WS_HB);
    unsigned short* afrag = (unsigned short*)(ws + WS_AFRAG);
    unsigned short* bfrag = (unsigned short*)(ws + WS_BFRAG);

    packW_kernel<<<DEPTH * NEXP * NNTILE, 256, 0, stream>>>(eW, bfrag);
    cycle_kernel<<<NTOK, DIM, 0, stream>>>(x, step, roots, projW, hA);

    float* cur = hA;
    float* nxt = hB;
    for (int l = 0; l < DEPTH; ++l) {
        gate_kernel<<<NTOK / 64, 64, 0, stream>>>(cur, gateW + (size_t)l * DIM * NEXP,
                                                  gateB + (size_t)l * NEXP, combine);
        packA_kernel<<<NTTILE, 256, 0, stream>>>(cur, afrag);
        moe_kernel<<<NSUPER * NNTILE, 256, 0, stream>>>(afrag,
                                                        bfrag + (size_t)l * BFRAG_LAYER_STRIDE,
                                                        combine, eb + (size_t)l * NEXP * DIM, nxt);
        ln_kernel<<<NTOK, 256, 0, stream>>>(nxt, gamma, beta);
        float* t = cur; cur = nxt; nxt = t;
    }
    head_kernel<<<BB, 256, 0, stream>>>(cur, headW, headB, out);
}
